// SparsifyWrapper_34170759807698
// MI455X (gfx1250) — matrix operand order repaired and checked by reading, not compiled, not run on hardware
//
#include <hip/hip_runtime.h>

#define N_TOK  4096
#define D_IN   768
#define D_SAE  32768
#define TOPK   64
#define MTILE  32            // token rows per workgroup
#define NCHUNK 256           // latent cols per chunk iteration (16 waves x 16)
#define KSTEPS (D_IN / 32)   // 24 WMMA k-steps of 32
#define NWAVES 16
#define BLOCK  512
#define NTILES (D_SAE / 16)  // 2048 column tiles
// packed B: [ntile][kstep][lane][16 bf16] = 2048*24*512 bf16 = 48 MB
#define PACKED_ELEMS ((size_t)NTILES * KSTEPS * 512)
#define PACKED_BYTES (PACKED_ELEMS * 2)

typedef __attribute__((ext_vector_type(16))) __bf16 v16bf;
typedef __attribute__((ext_vector_type(8)))  float  v8f;
typedef __attribute__((ext_vector_type(4)))  float  v4f;

// ---- one-time pack: W_enc f32 row-major -> bf16 WMMA B-fragment order ----
// B 32x16 frag order used here (contiguous k, matches A): lanes 0-15: N=lane, K=0..15 in elems; lanes 16-31: N=lane-16, K=16..31.
__global__ __launch_bounds__(BLOCK)
void pack_wenc(const float* __restrict__ W_enc, __bf16* __restrict__ Wp)
{
    const int gwave = (blockIdx.x * BLOCK + threadIdx.x) >> 5; // one wave per fragment
    const int lane  = threadIdx.x & 31;
    if (gwave >= NTILES * KSTEPS) return;
    const int t     = gwave % KSTEPS;
    const int ntile = gwave / KSTEPS;
    const float* src = W_enc + (size_t)(t * 32 + (lane >> 4) * 16) * D_SAE
                             + ntile * 16 + (lane & 15);
    v16bf b;
    #pragma unroll
    for (int e = 0; e < 16; ++e)
        b[e] = (__bf16)src[(size_t)e * D_SAE];
    *(v16bf*)(Wp + (size_t)gwave * 512 + lane * 16) = b;
}

template <bool PACKED>
__global__ __launch_bounds__(BLOCK, 1)
void sae_topk_fused(const float* __restrict__ x,
                    const float* __restrict__ W_enc,
                    const __bf16* __restrict__ Wp,
                    const float* __restrict__ b_enc,
                    const float* __restrict__ W_dec,
                    const float* __restrict__ b_dec,
                    float* __restrict__ out)
{
    // A fragments pre-swizzled to WMMA lane order: [kstep][mtile][lane][16 elems]
    __shared__ __attribute__((aligned(32))) __bf16 aFrag[KSTEPS * 2 * 32 * 16]; // 48 KB
    __shared__ float actsChunk[MTILE][NCHUNK + 4];                               // ~32.5 KB
    __shared__ float topv[MTILE][TOPK];                                          // 8 KB
    __shared__ int   topi[MTILE][TOPK];                                          // 8 KB

    const int tid     = threadIdx.x;
    const int lane    = tid & 31;
    const int wv      = tid >> 5;                 // wave id 0..15
    const int rowBase = blockIdx.x * MTILE;

    // ---- Stage A = bf16(x - b_dec) in WMMA fragment order ----
    // A 16x32 frag order used here (contiguous k, matches B): lanes 0-15: M=lane, K=0..15; lanes 16-31: M=lane-16, K=16..31.
    for (int i = tid; i < KSTEPS * 2 * 32 * 16; i += BLOCK) {
        int e  = i & 15;
        int L  = (i >> 4) & 31;
        int mt = (i >> 9) & 1;
        int t  = i >> 10;
        int m  = (L & 15) + mt * 16;
        int kk = t * 32 + ((L >> 4) << 4) + e;
        float v = x[(size_t)(rowBase + m) * D_IN + kk] - b_dec[kk];
        aFrag[i] = (__bf16)v;
    }
    for (int i = tid; i < MTILE * TOPK; i += BLOCK) {
        (&topv[0][0])[i] = 0.0f;
        (&topi[0][0])[i] = 0;
    }
    __syncthreads();

    float thr0 = 0.0f, thr1 = 0.0f;   // running 64th-largest per owned row

    for (int chunk = 0; chunk < D_SAE / NCHUNK; ++chunk) {
        const int n0g = chunk * NCHUNK + wv * 16 + (lane & 15);
        v8f acc0 = {}; v8f acc1 = {};

        const __bf16* fragBase = PACKED
            ? Wp + ((size_t)(chunk * 16 + wv) * KSTEPS) * 512 + lane * 16
            : nullptr;
        const float* bBase = PACKED
            ? nullptr
            : W_enc + (size_t)((lane >> 4) * 16) * D_SAE + n0g;

        if (PACKED && chunk + 1 < D_SAE / NCHUNK) {
            // warm near cache with next chunk's packed B region for this wave (24 KB)
            const char* nxt = (const char*)(Wp + ((size_t)((chunk + 1) * 16 + wv) * KSTEPS) * 512);
            #pragma unroll
            for (int p = 0; p < 6; ++p)
                __builtin_prefetch(nxt + (p * 32 + lane) * 128, 0, 1);
        }

        for (int t = 0; t < KSTEPS; ++t) {
            v16bf b;
            if (PACKED) {
                b = *(const v16bf*)(fragBase + (size_t)t * 512);  // 2x global_load_b128
            } else {
                const float* bp = bBase + (size_t)t * 32 * D_SAE;
                #pragma unroll
                for (int e = 0; e < 16; ++e)
                    b[e] = (__bf16)bp[(size_t)e * D_SAE];
            }
            v16bf a0 = *(const v16bf*)&aFrag[((t * 2 + 0) * 32 + lane) * 16];
            v16bf a1 = *(const v16bf*)&aFrag[((t * 2 + 1) * 32 + lane) * 16];

            acc0 = __builtin_amdgcn_wmma_f32_16x16x32_bf16(false, a0, false, b,
                                                           (short)0, acc0, false, false);
            acc1 = __builtin_amdgcn_wmma_f32_16x16x32_bf16(false, a1, false, b,
                                                           (short)0, acc1, false, false);
        }

        // bias + relu, spill chunk acts to LDS (D layout: vgpr r -> M = r + (lane>>4)*8)
        const float be   = b_enc[n0g];
        const int   colL = wv * 16 + (lane & 15);
        const int   rOff = (lane >> 4) << 3;
        #pragma unroll
        for (int r = 0; r < 8; ++r) {
            actsChunk[rOff + r][colL]      = fmaxf(acc0[r] + be, 0.0f);
            actsChunk[16 + rOff + r][colL] = fmaxf(acc1[r] + be, 0.0f);
        }
        __syncthreads();

        // ---- exact running top-64: wave wv owns rows 2wv, 2wv+1 ----
        #pragma unroll 1
        for (int rr = 0; rr < 2; ++rr) {
            const int r = wv * 2 + rr;
            float tcur = (rr == 0) ? thr0 : thr1;
            #pragma unroll 1
            for (int g = 0; g < NCHUNK / 32; ++g) {
                const int   c    = g * 32 + lane;
                const float v    = actsChunk[r][c];
                const int   gidx = chunk * NCHUNK + c;
                unsigned long long m = __ballot(v > tcur);
                while (m) {
                    const int src = __ffsll(m) - 1;
                    m &= m - 1;
                    const float cv = __shfl(v, src);
                    const int   ci = __shfl(gidx, src);
                    if (cv > tcur) {               // uniform branch (cv,tcur wave-uniform)
                        float e0 = topv[r][lane];
                        float e1 = topv[r][lane + 32];
                        float mv; int ms;
                        if (e1 < e0) { mv = e1; ms = lane + 32; } else { mv = e0; ms = lane; }
                        #pragma unroll
                        for (int off = 16; off; off >>= 1) {
                            float ov = __shfl_xor(mv, off);
                            int   os = __shfl_xor(ms, off);
                            if (ov < mv || (ov == mv && os < ms)) { mv = ov; ms = os; }
                        }
                        if (lane == 0) { topv[r][ms] = cv; topi[r][ms] = ci; }
                        float n0v = (ms == lane)      ? cv : e0;
                        float n1v = (ms == lane + 32) ? cv : e1;
                        float nm  = fminf(n0v, n1v);
                        #pragma unroll
                        for (int off = 16; off; off >>= 1)
                            nm = fminf(nm, __shfl_xor(nm, off));
                        tcur = nm;
                    }
                }
            }
            if (rr == 0) thr0 = tcur; else thr1 = tcur;
        }
        __syncthreads();
    }

    // ---- decode: recon = sum_k topv * W_dec[topi] + b_dec (float4 gather-FMA) ----
    #pragma unroll 1
    for (int rr = 0; rr < 2; ++rr) {
        const int r = wv * 2 + rr;
        v4f acc4[D_IN / 128];                       // 6 x float4 per lane
        #pragma unroll
        for (int j = 0; j < D_IN / 128; ++j) acc4[j] = (v4f){0.f, 0.f, 0.f, 0.f};

        #pragma unroll 1
        for (int k = 0; k < TOPK; ++k) {
            const float a  = topv[r][k];   // wave-uniform LDS broadcast
            const int   id = topi[r][k];
            if (a > 0.0f) {                // uniform branch
                const float* wd = W_dec + (size_t)id * D_IN + lane * 4;
                #pragma unroll
                for (int j = 0; j < D_IN / 128; ++j) {
                    v4f w = *(const v4f*)(wd + j * 128);
                    acc4[j] += a * w;
                }
            }
        }
        float* op = out + (size_t)(rowBase + r) * D_IN + lane * 4;
        const float* bd = b_dec + lane * 4;
        #pragma unroll
        for (int j = 0; j < D_IN / 128; ++j) {
            v4f b = *(const v4f*)(bd + j * 128);
            *(v4f*)(op + j * 128) = acc4[j] + b;
        }
    }
}

extern "C" void kernel_launch(void* const* d_in, const int* in_sizes, int n_in,
                              void* d_out, int out_size, void* d_ws, size_t ws_size,
                              hipStream_t stream) {
    const float* x     = (const float*)d_in[0];
    const float* W_enc = (const float*)d_in[1];
    const float* b_enc = (const float*)d_in[2];
    const float* W_dec = (const float*)d_in[3];
    const float* b_dec = (const float*)d_in[4];
    float* out = (float*)d_out;
    (void)in_sizes; (void)n_in; (void)out_size;

    dim3 grid(N_TOK / MTILE);   // 128 workgroups, 32 tokens each
    dim3 block(BLOCK);          // 16 waves (wave32)

    if (ws_size >= PACKED_BYTES) {
        __bf16* Wp = (__bf16*)d_ws;
        const int nfrag   = NTILES * KSTEPS;              // 49152 fragments
        const int nblocks = (nfrag * 32 + BLOCK - 1) / BLOCK;
        hipLaunchKernelGGL(pack_wenc, dim3(nblocks), block, 0, stream, W_enc, Wp);
        hipLaunchKernelGGL((sae_topk_fused<true>), grid, block, 0, stream,
                           x, W_enc, Wp, b_enc, W_dec, b_dec, out);
    } else {
        hipLaunchKernelGGL((sae_topk_fused<false>), grid, block, 0, stream,
                           x, W_enc, (const __bf16*)nullptr, b_enc, W_dec, b_dec, out);
    }
}
